// EventExtract_55722905699238
// MI455X (gfx1250) — matrix operand order repaired and checked by reading, not compiled, not run on hardware
//
#include <hip/hip_runtime.h>
#include <hip/hip_bf16.h>
#include <math.h>

typedef __attribute__((ext_vector_type(16))) _Float16 v16h;
typedef __attribute__((ext_vector_type(8)))  _Float16 v8h;
typedef __attribute__((ext_vector_type(8)))  float    v8f;

constexpr int Bn = 8, Nn = 32, Sn = 512, Hn = 768, Kn = 3, Cn = 128, Ln = 64;
constexpr int SP   = Sn + 2;        // zero-padded token rows
constexpr int KDIM = Hn * Kn;       // 2304
constexpr int NCH  = KDIM / 32;     // 72 K-chunks of 32
constexpr int NCT  = Cn / 16;       // 8 channel tiles
constexpr int NPT  = Sn / 16;       // 32 position tiles
constexpr int NS   = Bn * Nn;       // 256 spans

// ---------------------------------------------------------------- tokens -> f16 (zero padded)
__global__ void k_tok16(const float* __restrict__ tok, _Float16* __restrict__ o) {
  int i = blockIdx.x * 256 + threadIdx.x;        // b*SP*H + p*H + h
  int h = i % Hn; int bp = i / Hn; int p = bp % SP; int b = bp / SP;
  float v = (p < Sn) ? tok[((size_t)b * Sn + p) * Hn + h] : 0.f;
  o[i] = (_Float16)v;
}

// ------------------------------------------- pack 3 weight sets into WMMA B-fragment order
// packed[t][r0][ct][lane][e] ;  K_local = (e&7) + 16*(e/8) + 8*(lane/16), N = lane%16  (ISA 7.12.2, 32x16 B)
// reduction index r = k*H + h  =>  k-slices are contiguous chunk ranges (k0: 0..23, k1: 24..47)
__global__ void k_packw(const float* __restrict__ w0, const float* __restrict__ w1,
                        const float* __restrict__ w2, _Float16* __restrict__ o) {
  int i = blockIdx.x * 256 + threadIdx.x;        // 3*72*8*32*16 = 884736
  int e    = i & 15;
  int lane = (i >> 4) & 31;
  int ct   = (i >> 9) & 7;
  int rest = i >> 12;
  int r0   = rest % NCH;
  int t    = rest / NCH;
  int r = r0 * 32 + (e & 7) + ((e >> 3) << 4) + ((lane >= 16) ? 8 : 0);
  int c = ct * 16 + (lane & 15);
  int k = r / Hn, h = r % Hn;
  const float* w = (t == 0) ? w0 : ((t == 1) ? w1 : w2);
  o[i] = (_Float16)w[((size_t)c * Hn + h) * Kn + k];
}

// ------------------------------------------------- full-sequence conv maps via WMMA GEMM
// F[t][b][p][c] = sum_{k,h} tok[b,p+k,h] * w_t[c,h,k]   (zero-padded past S)
__global__ void k_conv(const _Float16* __restrict__ tok16, const _Float16* __restrict__ pw,
                       float* __restrict__ F) {
  int blk = blockIdx.x;                          // B*3*NPT = 768
  int pt  = blk & 31;
  int t   = (blk >> 5) % 3;
  int b   = blk / (3 * NPT);
  int tid  = threadIdx.x;
  int lane = tid & 31;
  int ct   = tid >> 5;                           // wave id = c-tile
  int m    = lane & 15;
  int hi   = lane >> 4;
  int p    = pt * 16 + m;

  const _Float16* tokb = tok16 + (size_t)b * SP * Hn;
  const _Float16* pwt  = pw + (size_t)t * NCH * (NCT * 32 * 16);

  v8f acc = {};
  for (int r0 = 0; r0 < NCH; ++r0) {
    int rbase = r0 * 32;
    int k  = rbase / Hn;
    int h0 = rbase % Hn;
    const _Float16* arow = tokb + (size_t)(p + k) * Hn + h0 + hi * 8;
    v8h alo = *(const v8h*)(arow);
    v8h ahi = *(const v8h*)(arow + 16);
    v16h a;
#pragma unroll
    for (int i = 0; i < 8; ++i) { a[i] = alo[i]; a[8 + i] = ahi[i]; }
    v16h bf = *(const v16h*)(pwt + ((size_t)(r0 * NCT + ct) * 32 + lane) * 16);
    acc = __builtin_amdgcn_wmma_f32_16x16x32_f16(false, a, false, bf, (short)0, acc,
                                                 false, false);
  }
  float* Fout = F + ((size_t)t * Bn + b) * Sn * Cn;
  int n = lane & 15;
#pragma unroll
  for (int r = 0; r < 8; ++r) {
    int prow = pt * 16 + r + 8 * hi;
    Fout[(size_t)prow * Cn + ct * 16 + n] = acc[r];
  }
}

// --------------------------------- running max: prefix map (in place), suffix map (-> Msuf)
// Two-level segment scan: 8 segments x 64 positions per (b, c) column.
__global__ void k_scan(float* __restrict__ F0, const float* __restrict__ F2,
                       float* __restrict__ Msuf) {
  int blk = blockIdx.x;                          // 2 maps * 8 b * 4 ctiles = 64
  int ct4 = blk & 3;
  int b   = (blk >> 2) & 7;
  int mp  = blk >> 5;                            // 0: prefix (ascending), 1: suffix (descending)
  int tid = threadIdx.x;                         // 256
  int cl  = tid & 31;
  int seg = tid >> 5;                            // 8 segments of 64
  int c   = ct4 * 32 + cl;
  __shared__ float smax[8][32];
  const float NEG = -3.4e38f;
  if (mp == 0) {
    float* base = F0 + (size_t)b * Sn * Cn + c;
    float mx = NEG;
    for (int j = 0; j < 64; ++j) mx = fmaxf(mx, base[(size_t)(seg * 64 + j) * Cn]);
    smax[seg][cl] = mx;
    __syncthreads();
    float carry = NEG;
    for (int s2 = 0; s2 < seg; ++s2) carry = fmaxf(carry, smax[s2][cl]);
    for (int j = 0; j < 64; ++j) {
      size_t idx = (size_t)(seg * 64 + j) * Cn;
      carry = fmaxf(carry, base[idx]);
      base[idx] = carry;                         // each thread owns its segment: no hazard
    }
  } else {
    const float* base = F2 + (size_t)b * Sn * Cn + c;
    float* obase = Msuf + (size_t)b * Sn * Cn + c;
    float mx = NEG;
    for (int j = 0; j < 64; ++j) mx = fmaxf(mx, base[(size_t)(seg * 64 + j) * Cn]);
    smax[seg][cl] = mx;
    __syncthreads();
    float carry = NEG;
    for (int s2 = seg + 1; s2 < 8; ++s2) carry = fmaxf(carry, smax[s2][cl]);
    for (int j = 63; j >= 0; --j) {
      size_t idx = (size_t)(seg * 64 + j) * Cn;
      carry = fmaxf(carry, base[idx]);
      obase[idx] = carry;
    }
  }
}

// --------------------------- gather the 4 boundary token rows per span -> G[s][4][768] f16
// r0: start-2, r1: start-1, r2: end-2, r3: end-1  (zero row when index < 0)
__global__ void k_gather(const float* __restrict__ tok, const int* __restrict__ spans,
                         _Float16* __restrict__ G) {
  int i = blockIdx.x * 256 + threadIdx.x;        // 256*4*768 = 786432
  int h = i % Hn;
  int sr = i / Hn;
  int r = sr & 3;
  int s = sr >> 2;
  int b = s >> 5;
  int start = spans[s * 2], end = spans[s * 2 + 1];
  int row = (r == 0) ? start - 2 : (r == 1) ? start - 1 : (r == 2) ? end - 2 : end - 1;
  float v = (row >= 0) ? tok[((size_t)b * Sn + row) * Hn + h] : 0.f;
  G[i] = (_Float16)v;
}

// ---------------- partial boundary windows as 4 small WMMA GEMMs -> P[job][s][c] f32
// job0: pre2 = G[r0]*Wpre_k0 + G[r1]*Wpre_k1   (K=1536, B chunks 0..47 of t=0)
// job1: pre1 = G[r1]*Wpre_k0                   (K=768,  B chunks 0..23 of t=0)
// job2: spn2 = G[r2]*Wspn_k0 + G[r3]*Wspn_k1   (K=1536, t=1)
// job3: spn1 = G[r3]*Wspn_k0                   (K=768,  t=1)
__global__ void k_part(const _Float16* __restrict__ G, const _Float16* __restrict__ pw,
                       float* __restrict__ P) {
  const int roffs[4] = {0, 1, 2, 3};
  const int nchs[4]  = {48, 24, 48, 24};
  const int tsel[4]  = {0, 0, 1, 1};
  int blk = blockIdx.x;                          // 4 jobs * 16 m-tiles = 64
  int mt  = blk & 15;
  int job = blk >> 4;
  int tid  = threadIdx.x;
  int lane = tid & 31;
  int ct   = tid >> 5;
  int m    = lane & 15;
  int hi   = lane >> 4;
  int s    = mt * 16 + m;

  const _Float16* aBase = G + (size_t)s * (4 * Hn) + roffs[job] * Hn;
  const _Float16* pwt   = pw + (size_t)tsel[job] * NCH * (NCT * 32 * 16);
  int nch = nchs[job];

  v8f acc = {};
  for (int r0 = 0; r0 < nch; ++r0) {
    const _Float16* arow = aBase + r0 * 32 + hi * 8;
    v8h alo = *(const v8h*)(arow);
    v8h ahi = *(const v8h*)(arow + 16);
    v16h a;
#pragma unroll
    for (int i = 0; i < 8; ++i) { a[i] = alo[i]; a[8 + i] = ahi[i]; }
    v16h bf = *(const v16h*)(pwt + ((size_t)(r0 * NCT + ct) * 32 + lane) * 16);
    acc = __builtin_amdgcn_wmma_f32_16x16x32_f16(false, a, false, bf, (short)0, acc,
                                                 false, false);
  }
  float* Pout = P + (size_t)job * NS * Cn;
  int n = lane & 15;
#pragma unroll
  for (int r = 0; r < 8; ++r) {
    int srow = mt * 16 + r + 8 * hi;
    Pout[(size_t)srow * Cn + ct * 16 + n] = acc[r];
  }
}

// ------------------------------------- per-span: O(1) lookups + short span max + FC + sigmoid
__global__ void k_span(const float* __restrict__ F, const float* __restrict__ Msuf,
                       const float* __restrict__ P,
                       const float* __restrict__ bpre, const float* __restrict__ bspn,
                       const float* __restrict__ bsuf,
                       const float* __restrict__ fcw, const float* __restrict__ fcb,
                       const int* __restrict__ spans,
                       float* __restrict__ out_probs, int* __restrict__ out_pred) {
  __shared__ float feats[3 * Cn];
  int s = blockIdx.x;
  int b = s >> 5;
  int c = threadIdx.x;                           // 128 threads
  int start = spans[s * 2], end = spans[s * 2 + 1];
  int w = end - start;
  const float NEG = -3.4e38f;

  // ---- prefix: windows t in [0,493]
  {
    float m = (start <= 493) ? 0.f : NEG;
    if (start >= 3)
      m = fmaxf(m, F[(((size_t)0 * Bn + b) * Sn + start - 3) * Cn + c]);  // cummax[0..start-3]
    if (start >= 2 && start <= 495) m = fmaxf(m, P[((size_t)0 * NS + s) * Cn + c]);
    if (start >= 1 && start <= 494) m = fmaxf(m, P[((size_t)1 * NS + s) * Cn + c]);
    feats[c] = bpre[c] + m;
  }
  // ---- span: windows t in [0,13]
  {
    const float* Fs = F + ((size_t)1 * Bn + b) * Sn * Cn + c;
    float m = (w <= 13) ? 0.f : NEG;
    for (int p = start; p <= start + w - 3; ++p) m = fmaxf(m, Fs[(size_t)p * Cn]);
    if (w >= 2 && w <= 15) m = fmaxf(m, P[((size_t)2 * NS + s) * Cn + c]);
    if (w <= 14) m = fmaxf(m, P[((size_t)3 * NS + s) * Cn + c]);
    feats[Cn + c] = bspn[c] + m;
  }
  // ---- suffix: p in [end, min(end+508,511)]
  {
    float m;
    if (end >= 3) {
      m = (end >= 4) ? 0.f : NEG;
      m = fmaxf(m, Msuf[((size_t)b * Sn + end) * Cn + c]);  // cummax[end..511]
    } else {                                     // end in {1,2}: exact range on raw map
      const float* Fu = F + ((size_t)2 * Bn + b) * Sn * Cn + c;
      m = NEG;
      int phi = end + 508;
      for (int p = end; p <= phi; ++p) m = fmaxf(m, Fu[(size_t)p * Cn]);
    }
    feats[2 * Cn + c] = bsuf[c] + m;
  }
  __syncthreads();
  if (c < Ln) {
    float acc = fcb[c];
    const float* fw = fcw + (size_t)c * 3 * Cn;
    for (int j = 0; j < 3 * Cn; ++j) acc += feats[j] * fw[j];
    float pr = 1.f / (1.f + expf(-acc));
    out_probs[s * Ln + c] = pr;
    out_pred[s * Ln + c] = (pr > 0.5f) ? 1 : 0;
  }
}

extern "C" void kernel_launch(void* const* d_in, const int* in_sizes, int n_in,
                              void* d_out, int out_size, void* d_ws, size_t ws_size,
                              hipStream_t stream) {
  (void)in_sizes; (void)n_in; (void)out_size; (void)ws_size;
  const float* tokens   = (const float*)d_in[0];
  const float* span_w   = (const float*)d_in[1];
  const float* span_b   = (const float*)d_in[2];
  const float* prefix_w = (const float*)d_in[3];
  const float* prefix_b = (const float*)d_in[4];
  const float* suffix_w = (const float*)d_in[5];
  const float* suffix_b = (const float*)d_in[6];
  const float* fc_w     = (const float*)d_in[7];
  const float* fc_b     = (const float*)d_in[8];
  const int*   spans    = (const int*)d_in[9];

  char* ws = (char*)d_ws;
  size_t off = 0;
  _Float16* tok16 = (_Float16*)(ws + off); off += (size_t)Bn * SP * Hn * 2;            // 6.0 MB
  _Float16* packw = (_Float16*)(ws + off); off += (size_t)3 * NCH * NCT * 32 * 16 * 2; // 1.7 MB
  float*    Fmap  = (float*)(ws + off);    off += (size_t)3 * Bn * Sn * Cn * 4;        // 6.0 MB
  float*    Msuf  = (float*)(ws + off);    off += (size_t)Bn * Sn * Cn * 4;            // 2.0 MB
  _Float16* G     = (_Float16*)(ws + off); off += (size_t)NS * 4 * Hn * 2;             // 1.5 MB
  float*    P     = (float*)(ws + off);    off += (size_t)4 * NS * Cn * 4;             // 0.5 MB

  float* F0 = Fmap;                              // prefix map (becomes cummax in place)
  float* F2 = Fmap + (size_t)2 * Bn * Sn * Cn;   // suffix map (stays raw)

  k_tok16<<<(Bn * SP * Hn) / 256, 256, 0, stream>>>(tokens, tok16);
  k_packw<<<(3 * NCH * NCT * 32 * 16) / 256, 256, 0, stream>>>(prefix_w, span_w, suffix_w, packw);
  k_gather<<<(NS * 4 * Hn) / 256, 256, 0, stream>>>(tokens, spans, G);
  k_conv<<<Bn * 3 * NPT, 256, 0, stream>>>(tok16, packw, Fmap);
  k_part<<<4 * (NS / 16), 256, 0, stream>>>(G, packw, P);
  k_scan<<<64, 256, 0, stream>>>(F0, F2, Msuf);
  k_span<<<NS, 128, 0, stream>>>(Fmap, Msuf, P, prefix_b, span_b, suffix_b,
                                 fc_w, fc_b, spans,
                                 (float*)d_out, ((int*)d_out) + NS * Ln);
}
